// MyFullyActiveRQS_72258529788183
// MI455X (gfx1250) — matrix operand order repaired and checked by reading, not compiled, not run on hardware
//
#include <hip/hip_runtime.h>
#include <hip/hip_bf16.h>
#include <cstdint>

// ---------------------------------------------------------------------------
// MI455X (gfx1250) implementation.
// conv3 (3x3, 128->276) dominates (~83 of ~93 GFLOP); conv inputs are bf16 and
// L2-resident (192MB L2), so the stack is WMMA-issue bound. Implicit GEMM on
// v_wmma_f32_16x16x32_bf16, tap-major k-loop (no runtime division in the hot
// path), double-buffered LDS for A, B fragments loaded DIRECTLY from global in
// pre-swizzled fragment order (weights are L2-resident), zero-guard buffer
// instead of per-element border masking.
// ---------------------------------------------------------------------------

typedef __attribute__((ext_vector_type(16))) __bf16 v16bf;
typedef __attribute__((ext_vector_type(8)))  float  v8f;

#define BATCH   32
#define CCH     12
#define HWX     4096        // 64*64
#define CHWX    (CCH*HWX)   // 49152
#define MPIX    (BATCH*HWX) // 131072 pixels = GEMM M
#define NPARAM  23          // 3*8-1 spline params per channel
#define OUTC    (CCH*NPARAM) // 276
#define TAILV   3.0f
#define MBWc    0.001f
#define MBHc    0.001f
#define MDVc    0.001f

union Frag { v16bf v; uint4 q[2]; };

// ---------------------------------------------------------------------------
// Pack concat(x, conditioning, zero-pad) NCHW f32 -> bf16 [B,32,H,W]
// (Cin padded 16 -> 32 so conv1's K per tap is one full WMMA-K tile.)
// ---------------------------------------------------------------------------
__global__ __launch_bounds__(256)
void pack_input_kernel(const float* __restrict__ x, const float* __restrict__ cond,
                       __bf16* __restrict__ out) {
    int i = blockIdx.x * 256 + threadIdx.x;       // total 32*32*4096
    int b = i >> 17;                              // / (32*4096)
    int c = (i >> 12) & 31;
    int p = i & 4095;
    float v = 0.f;
    if (c < CCH)       v = x[((size_t)(b * CCH + c) << 12) + p];
    else if (c < 16)   v = cond[((size_t)(b * 4 + (c - CCH)) << 12) + p];
    out[i] = (__bf16)v;
}

// ---------------------------------------------------------------------------
// OIHW f32 weights -> bf16 in B-FRAGMENT order: wp[((kt*Npad + n)<<5) + kk]
// where k_linear = kt*32+kk = tap*Cinp + ic (tap-major), zero padded in
// ic (Cin->Cinp) and n (Nreal->Npad). GEMM waves then load B fragments
// straight from global (L2-resident) with two b128 loads per fragment.
// ---------------------------------------------------------------------------
__global__ __launch_bounds__(256)
void pack_weights_kernel(const float* __restrict__ w, __bf16* __restrict__ wp,
                         int CinReal, int Cinp, int KHKW, int Nreal, int Npad,
                         int total) {
    int i = blockIdx.x * 256 + threadIdx.x;
    if (i >= total) return;
    int kk  = i & 31;
    int n32 = i >> 5;              // kt*Npad + n
    int kt  = n32 / Npad;
    int n   = n32 - kt * Npad;
    int kl  = kt * 32 + kk;        // tap-major linear k
    int tap = kl / Cinp;
    int ic  = kl - tap * Cinp;
    float v = 0.f;
    if (ic < CinReal && n < Nreal)
        v = w[(size_t)(n * CinReal + ic) * KHKW + tap];
    wp[i] = (__bf16)v;
}

// ---------------------------------------------------------------------------
// Zero-guard buffer fill (border pixels redirect their A loads here).
// ---------------------------------------------------------------------------
__global__ __launch_bounds__(256)
void zero_fill_kernel(uint4* __restrict__ p) {
    p[blockIdx.x * 256 + threadIdx.x] = uint4{0u, 0u, 0u, 0u};
}

// ---------------------------------------------------------------------------
// Implicit-GEMM conv, SAME padding, NCHW, H=W=64, M=131072 pixels.
// Block tile 128(M) x (2*FN*16)(N) x 32(K); 8 waves = 4(M) x 2(N);
// each wave: 32 x FN*16 = 2 x FN WMMA fragments.
// k-loop tap-major; pixel geometry loop-invariant; A double-buffered in LDS;
// B fragments loaded directly from global (pre-swizzled, L2-resident).
// EPI==0: bias+ReLU -> bf16 NCHW (one b128 store per frag-row).
// EPI==1: bias -> f32 params NCHW, guarded n < Nreal (two float4 stores).
// ---------------------------------------------------------------------------
template<int EPI, int KH, int KW, int CTILES, int FN>
__global__ __launch_bounds__(256)
void conv_gemm_kernel(const __bf16* __restrict__ src, const __bf16* __restrict__ wp,
                      const float* __restrict__ bias, float* __restrict__ dstf,
                      __bf16* __restrict__ dstb, const __bf16* __restrict__ zb,
                      int Nreal, int Npad) {
    constexpr int TAPS = KH * KW;
    constexpr int KT   = TAPS * CTILES;
    constexpr int CINP = 32 * CTILES;
    constexpr int BN   = 2 * FN * 16;

    __shared__ __align__(16) __bf16 As[2][128][40];  // [buf][m][k], +8 pad

    const int tid  = threadIdx.x;
    const int m0   = blockIdx.x << 7;
    const int n0   = blockIdx.y * BN;
    const int wave = tid >> 5, lane = tid & 31;
    const int wm   = (wave >> 1) << 5;       // 0,32,64,96
    const int wn   = (wave & 1) * (FN * 16); // n sub-block
    const int lm   = lane & 15;
    const int half = lane >> 4;

    // ---- A-feeder thread-invariant geometry (hoisted out of k-loop) ----
    const int amm  = tid & 127;          // LDS row (pixel within tile)
    const int akh  = (tid >> 7) << 4;    // k sub-offset: 0 or 16
    const int am   = m0 + amm;
    const int ab   = am >> 12, apx = am & 4095;
    const int ay   = apx >> 6,  ax  = apx & 63;
    const size_t achan = ((size_t)ab * CINP) << 12;

    union AV { __bf16 h[16]; uint4 q[2]; };

    auto loadA = [&](int kt, AV& a) {
        const int tap = kt / CTILES;                 // constexpr divisor
        const int ct  = kt - tap * CTILES;
        const int dy  = tap / KW, dx = tap - (tap / KW) * KW;
        const int iy  = ay + dy - (KH >> 1);
        const int ix  = ax + dx - (KW >> 1);
        const bool valid = ((unsigned)iy < 64u) & ((unsigned)ix < 64u);
        const __bf16* sp = valid
            ? src + achan + ((size_t)(ct * 32 + akh) << 12) + ((iy << 6) + ix)
            : zb;                                    // zero-guard: loads give 0
        #pragma unroll
        for (int i = 0; i < 16; ++i)                 // const ioffset loads
            a.h[i] = sp[(size_t)i << 12];
    };
    auto storeA = [&](int buf, const AV& a) {
        *(uint4*)&As[buf][amm][akh]     = a.q[0];
        *(uint4*)&As[buf][amm][akh + 8] = a.q[1];
    };
    // B fragments straight from global, pre-swizzled fragment order.
    auto loadB = [&](int kt, Frag (&bf)[FN]) {
        const __bf16* base = wp + (((size_t)kt * Npad + n0 + wn + lm) << 5)
                                + half * 8;
        #pragma unroll
        for (int f = 0; f < FN; ++f) {
            bf[f].q[0] = *(const uint4*)(base + (f << 9));      // n += 16
            bf[f].q[1] = *(const uint4*)(base + (f << 9) + 16);
        }
    };

    v8f acc[2][FN];
    #pragma unroll
    for (int fm = 0; fm < 2; ++fm)
        #pragma unroll
        for (int fn = 0; fn < FN; ++fn) acc[fm][fn] = v8f{};

    {   // prologue: A tile 0 into buffer 0
        AV a0v;
        loadA(0, a0v);
        storeA(0, a0v);
    }

    int buf = 0;
    #pragma unroll 2
    for (int kt = 0; kt < KT; ++kt) {
        __syncthreads();

        AV an;
        if (kt + 1 < KT) loadA(kt + 1, an);           // overlap with WMMAs
        if (kt + 2 < KT)                              // weight stream prefetch
            __builtin_prefetch(wp + (((size_t)(kt + 2) * Npad + n0 + wn + lm) << 5), 0, 1);

        Frag bf[FN];
        loadB(kt, bf);

        // A fragments per gfx1250 16-bit layout:
        // lanes 0-15 k={0..7,16..23}; lanes 16-31 k={8..15,24..31}
        Frag a0, a1;
        a0.q[0] = *(const uint4*)&As[buf][wm + lm][half * 8];
        a0.q[1] = *(const uint4*)&As[buf][wm + lm][16 + half * 8];
        a1.q[0] = *(const uint4*)&As[buf][wm + 16 + lm][half * 8];
        a1.q[1] = *(const uint4*)&As[buf][wm + 16 + lm][16 + half * 8];

        #pragma unroll
        for (int fn = 0; fn < FN; ++fn) {
            acc[0][fn] = __builtin_amdgcn_wmma_f32_16x16x32_bf16(
                false, a0.v, false, bf[fn].v, (short)0, acc[0][fn], false, false);
            acc[1][fn] = __builtin_amdgcn_wmma_f32_16x16x32_bf16(
                false, a1.v, false, bf[fn].v, (short)0, acc[1][fn], false, false);
        }

        if (kt + 1 < KT) storeA(buf ^ 1, an);
        buf ^= 1;
    }

    // ---- Epilogue: C/D layout lane L -> n=L&15, vgpr r -> m=r+(L>=16)*8.
    // The 8 rows per accumulator are 8 contiguous NCHW pixels -> vector store.
    #pragma unroll
    for (int fm = 0; fm < 2; ++fm) {
        #pragma unroll
        for (int fn = 0; fn < FN; ++fn) {
            v8f a = acc[fm][fn];
            int n = n0 + wn + fn * 16 + lm;
            if (n >= Nreal) continue;
            float bz = bias[n];
            int mb = m0 + wm + fm * 16 + half * 8;   // 8-aligned pixel base
            int b  = mb >> 12, p = mb & 4095;
            if (EPI == 0) {
                union { __bf16 h[8]; uint4 q; } ov;
                #pragma unroll
                for (int r = 0; r < 8; ++r) {
                    float v = a[r] + bz;
                    ov.h[r] = (__bf16)(v > 0.f ? v : 0.f);
                }
                *(uint4*)&dstb[((size_t)(b * Nreal + n) << 12) + p] = ov.q;
            } else {
                float* dp = dstf + ((size_t)(b * Nreal + n) << 12) + p;
                *(float4*)dp       = make_float4(a[0] + bz, a[1] + bz,
                                                 a[2] + bz, a[3] + bz);
                *(float4*)(dp + 4) = make_float4(a[4] + bz, a[5] + bz,
                                                 a[6] + bz, a[7] + bz);
            }
        }
    }
}

// ---------------------------------------------------------------------------
// logdet init (copy input logdet; rqs blocks atomically accumulate on top)
// ---------------------------------------------------------------------------
__global__ void init_logdet_kernel(const float* __restrict__ ld, float* __restrict__ out) {
    int i = threadIdx.x;
    if (i < BATCH) out[i] = ld[i];
}

// ---------------------------------------------------------------------------
// RQS spline: register-resident; cndmask-chain bin gather (no scratch).
// One 256-thread block == one batch slice (49152 % 256 == 0).
// ---------------------------------------------------------------------------
__global__ __launch_bounds__(256)
void rqs_kernel(const float* __restrict__ xg, const float* __restrict__ pg,
                float* __restrict__ yout, float* __restrict__ ldout) {
    int e   = blockIdx.x * 256 + threadIdx.x;
    int b   = e / CHWX;
    int rem = e - b * CHWX;
    int c   = rem >> 12;
    int p   = rem & 4095;
    const float* P = pg + ((size_t)b * OUTC + c * NPARAM) * (size_t)HWX + p;

    float uw[8], uh[8], d[9], cw[9], ch[9], wdt[8], hgt[8];
    #pragma unroll
    for (int j = 0; j < 8; ++j) uw[j] = P[(size_t)j * HWX];
    #pragma unroll
    for (int j = 0; j < 8; ++j) uh[j] = P[(size_t)(8 + j) * HWX];

    // derivs: pad value log(exp(1-MD)-1) => softplus gives exactly 1-MD
    d[0] = 1.0f; d[8] = 1.0f;
    #pragma unroll
    for (int j = 0; j < 7; ++j) {
        float u  = P[(size_t)(16 + j) * HWX];
        float sp = (u > 20.f) ? u : log1pf(__expf(u));
        d[j + 1] = MDVc + sp;
    }

    // widths: softmax -> knots
    float mx = uw[0];
    #pragma unroll
    for (int j = 1; j < 8; ++j) mx = fmaxf(mx, uw[j]);
    float s = 0.f;
    #pragma unroll
    for (int j = 0; j < 8; ++j) { uw[j] = __expf(uw[j] - mx); s += uw[j]; }
    float inv = 1.f / s, run = 0.f;
    cw[0] = -TAILV;
    #pragma unroll
    for (int j = 0; j < 8; ++j) {
        run += MBWc + (1.f - 8.f * MBWc) * uw[j] * inv;
        cw[j + 1] = 2.f * TAILV * run - TAILV;
    }
    cw[8] = TAILV;
    #pragma unroll
    for (int j = 0; j < 8; ++j) wdt[j] = cw[j + 1] - cw[j];

    // heights
    mx = uh[0];
    #pragma unroll
    for (int j = 1; j < 8; ++j) mx = fmaxf(mx, uh[j]);
    s = 0.f;
    #pragma unroll
    for (int j = 0; j < 8; ++j) { uh[j] = __expf(uh[j] - mx); s += uh[j]; }
    inv = 1.f / s; run = 0.f;
    ch[0] = -TAILV;
    #pragma unroll
    for (int j = 0; j < 8; ++j) {
        run += MBHc + (1.f - 8.f * MBHc) * uh[j] * inv;
        ch[j + 1] = 2.f * TAILV * run - TAILV;
    }
    ch[8] = TAILV;
    #pragma unroll
    for (int j = 0; j < 8; ++j) hgt[j] = ch[j + 1] - ch[j];

    float xv = xg[e];
    bool inside = (xv >= -TAILV) && (xv <= TAILV);
    float xin = fminf(fmaxf(xv, -TAILV), TAILV);
    int idx = 0;
    #pragma unroll
    for (int j = 1; j < 8; ++j) idx += (xin >= cw[j]) ? 1 : 0;   // 0..7

    float icw = cw[0], iw = wdt[0], ich = ch[0], ih = hgt[0], d0 = d[0], d1 = d[1];
    #pragma unroll
    for (int j = 1; j < 8; ++j) {
        bool sel = (j == idx);
        icw = sel ? cw[j]  : icw;
        iw  = sel ? wdt[j] : iw;
        ich = sel ? ch[j]  : ich;
        ih  = sel ? hgt[j] : ih;
        d0  = sel ? d[j]   : d0;
        d1  = sel ? d[j+1] : d1;
    }

    float idel  = ih / iw;
    float th    = (xin - icw) / iw;
    float tom   = th * (1.f - th);
    float numer = ih * (idel * th * th + d0 * tom);
    float denom = idel + (d0 + d1 - 2.f * idel) * tom;
    float yv    = ich + numer / denom;
    float omt   = 1.f - th;
    float dnum  = idel * idel * (d1 * th * th + 2.f * idel * tom + d0 * omt * omt);
    float lad   = logf(dnum) - 2.f * logf(denom);

    yv  = inside ? yv  : xv;
    lad = inside ? lad : 0.f;
    yout[e] = yv;

    __shared__ float red[256];
    red[threadIdx.x] = lad;
    __syncthreads();
    #pragma unroll
    for (int s2 = 128; s2 > 0; s2 >>= 1) {
        if (threadIdx.x < s2) red[threadIdx.x] += red[threadIdx.x + s2];
        __syncthreads();
    }
    if (threadIdx.x == 0) atomicAdd(&ldout[b], red[0]);
}

// ---------------------------------------------------------------------------
extern "C" void kernel_launch(void* const* d_in, const int* in_sizes, int n_in,
                              void* d_out, int out_size, void* d_ws, size_t ws_size,
                              hipStream_t stream) {
    const float* x    = (const float*)d_in[0];
    const float* ld   = (const float*)d_in[1];
    const float* cond = (const float*)d_in[2];
    const float* w1   = (const float*)d_in[3];
    const float* b1   = (const float*)d_in[4];
    const float* w2   = (const float*)d_in[5];
    const float* b2   = (const float*)d_in[6];
    const float* w3   = (const float*)d_in[7];
    const float* b3   = (const float*)d_in[8];
    float* out = (float*)d_out;

    // Workspace layout (bytes): fully rewritten each call (graph-safe).
    char* ws = (char*)d_ws;
    __bf16* netin  = (__bf16*)(ws);                          //  8 MiB: [32,32,64,64]
    __bf16* h1     = (__bf16*)(ws + (size_t)8   * 1048576);  // 32 MiB: [32,128,64,64]
    __bf16* h2     = (__bf16*)(ws + (size_t)40  * 1048576);  // 32 MiB
    float*  params = (float* )(ws + (size_t)72  * 1048576);  // 138 MiB: [32,276,64,64]
    __bf16* w1p    = (__bf16*)(ws + (size_t)212 * 1048576);  // 9 kt  x 128 x 32
    __bf16* w2p    = (__bf16*)(ws + (size_t)214 * 1048576);  // 4 kt  x 128 x 32
    __bf16* w3p    = (__bf16*)(ws + (size_t)216 * 1048576);  // 36 kt x 288 x 32
    __bf16* zbuf   = (__bf16*)(ws + (size_t)218 * 1048576);  // 128 KiB zeros

    // 1) pack inputs / weights to bf16 (fragment-order B layout), zero guard
    pack_input_kernel<<<(BATCH * 32 * HWX) / 256, 256, 0, stream>>>(x, cond, netin);
    pack_weights_kernel<<<(9 * 128 * 32 + 255) / 256, 256, 0, stream>>>(
        w1, w1p, 16, 32, 9, 128, 128, 9 * 128 * 32);
    pack_weights_kernel<<<(4 * 128 * 32 + 255) / 256, 256, 0, stream>>>(
        w2, w2p, 128, 128, 1, 128, 128, 4 * 128 * 32);
    pack_weights_kernel<<<(36 * 288 * 32 + 255) / 256, 256, 0, stream>>>(
        w3, w3p, 128, 128, 9, 276, 288, 36 * 288 * 32);
    zero_fill_kernel<<<32, 256, 0, stream>>>((uint4*)zbuf);   // 128 KiB

    // 2) conv stack as implicit WMMA GEMMs (M = 131072 pixels)
    dim3 blk(256);
    conv_gemm_kernel<0, 3, 3, 1, 2><<<dim3(MPIX / 128, 2), blk, 0, stream>>>(
        netin, w1p, b1, nullptr, h1, zbuf, 128, 128);
    conv_gemm_kernel<0, 1, 1, 4, 2><<<dim3(MPIX / 128, 2), blk, 0, stream>>>(
        h1, w2p, b2, nullptr, h2, zbuf, 128, 128);
    conv_gemm_kernel<1, 3, 3, 4, 3><<<dim3(MPIX / 128, 3), blk, 0, stream>>>(
        h2, w3p, b3, params, nullptr, zbuf, 276, 288);

    // 3) spline + logdet
    init_logdet_kernel<<<1, 32, 0, stream>>>(ld, out + (size_t)BATCH * CHWX);
    rqs_kernel<<<(BATCH * CHWX) / 256, 256, 0, stream>>>(
        x, params, out, out + (size_t)BATCH * CHWX);
}
